// Mamba_5738076307810
// MI455X (gfx1250) — matrix operand order repaired and checked by reading, not compiled, not run on hardware
//
#include <hip/hip_runtime.h>
#include <hip/hip_bf16.h>

typedef __bf16 bf16_t;
typedef __attribute__((ext_vector_type(16))) __bf16 bf16x16;
typedef __attribute__((ext_vector_type(8)))  __bf16 bf16x8;
typedef __attribute__((ext_vector_type(8)))  float  f32x8;
typedef __attribute__((ext_vector_type(4)))  unsigned int u32x4;
typedef __attribute__((ext_vector_type(8)))  int i32x8;
typedef __attribute__((ext_vector_type(4)))  int i32x4;

#define BATCH  2
#define SEQ    2048
#define DMODEL 1024
#define DINNER 2048
#define NSTATE 16
#define CONVW  4
#define RRANK  64
#define MROWS  (BATCH * SEQ)        /* 4096 */
#define XPAD   128                   /* x_proj output padded 96 -> 128 */

/* ---------------- bf16 WMMA GEMM:  C[M,N] = A[M,K] * W[N,K]^T (+bias, act) -------- */
#define BM 64
#define BN 128
#define BK 32
#define LDT (BK + 8)   /* 40 halfs = 80 B rows; TDM pad: +4 DWORDs every 16 DWORDs */

#if defined(__has_builtin)
#if __has_builtin(__builtin_amdgcn_tensor_load_to_lds) && __has_builtin(__builtin_amdgcn_s_wait_tensorcnt)
#define HAVE_TDM 1
#endif
#endif
#ifndef HAVE_TDM
#define HAVE_TDM 0
#endif

__device__ __forceinline__ unsigned lds_off(const void* p) {
    /* LDS aperture: low 32 bits of a flat shared address are the LDS byte offset */
    return (unsigned)(uintptr_t)p;
}

#if HAVE_TDM
/* Issue one TDM 2D tile load: rows x BK bf16 elements from row-major global
 * (row stride = K elements) into LDS at lds_byte_off, padded to LDT-half rows. */
__device__ __forceinline__ void tdm_issue_tile(const bf16_t* gptr, unsigned lds_byte_off,
                                               int rows, int K) {
    unsigned long long ga = (unsigned long long)(uintptr_t)gptr;
    u32x4 g0;
    g0[0] = 1u;                                            /* count=1, user mode      */
    g0[1] = lds_byte_off;                                  /* lds_addr                */
    g0[2] = (unsigned)(ga & 0xFFFFFFFFu);                  /* global_addr[31:0]       */
    g0[3] = (unsigned)((ga >> 32) & 0x01FFFFFFu) | (2u << 30); /* addr[56:32], type=2 */

    unsigned td0 = (unsigned)K;      /* tensor_dim0 (elements)           */
    unsigned td1 = (unsigned)rows;   /* tensor_dim1                      */
    i32x8 g1;
    /* data_size=1(2B) | pad_enable | pad_interval=3(16 DW) | pad_amount=3(4 DW) */
    g1[0] = (int)((1u << 16) | (1u << 20) | (3u << 22) | (3u << 25));
    g1[1] = (int)((td0 & 0xFFFFu) << 16);                    /* tensor_dim0[15:0]   */
    g1[2] = (int)((td0 >> 16) | ((td1 & 0xFFFFu) << 16));    /* dim0 hi | dim1 lo   */
    g1[3] = (int)((td1 >> 16) | ((unsigned)BK << 16));       /* dim1 hi | tile_dim0 */
    g1[4] = (int)rows;                                       /* tile_dim1, tile_dim2=0 */
    g1[5] = (int)(unsigned)K;                                /* tensor_dim0_stride lo  */
    g1[6] = 0;                                               /* stride hi | dim1_stride lo */
    g1[7] = 0;
    i32x4 z4 = {0, 0, 0, 0};
    i32x8 z8 = {0, 0, 0, 0, 0, 0, 0, 0};
    __builtin_amdgcn_tensor_load_to_lds(g0, g1, z4, z4, z8, 0);
}
#endif

__device__ __forceinline__ bf16x16 frag_a(const bf16_t* sA, int row) {
    int lane = threadIdx.x & 31;
    int m    = lane & 15;
    int khi  = (lane & 16) ? 8 : 0;
    const bf16_t* p = sA + (row + m) * LDT;
    bf16x8 lo = *(const bf16x8*)(p + khi);
    bf16x8 hi = *(const bf16x8*)(p + 16 + khi);
    bf16x16 f;
#pragma unroll
    for (int h = 0; h < 8; ++h) { f[h] = lo[h]; f[h + 8] = hi[h]; }
    return f;
}

__device__ __forceinline__ bf16x16 frag_b(const bf16_t* sB, int col) {
    int lane = threadIdx.x & 31;
    int n    = lane & 15;
    int kb   = (lane & 16) ? 8 : 0;
    const bf16_t* p = sB + (col + n) * LDT + kb;
    bf16x8 lo = *(const bf16x8*)(p);
    bf16x8 hi = *(const bf16x8*)(p + 16);
    bf16x16 f;
#pragma unroll
    for (int h = 0; h < 8; ++h) { f[h] = lo[h]; f[h + 8] = hi[h]; }
    return f;
}

__global__ void __launch_bounds__(256)
gemm_bf16_nt(const bf16_t* __restrict__ A, const bf16_t* __restrict__ W,
             float* __restrict__ C, const float* __restrict__ bias,
             int M, int N, int K, int act /*0=none,1=softplus*/)
{
    __shared__ bf16_t sA[2][BM * LDT];
    __shared__ bf16_t sB[2][BN * LDT];

    const int tid  = threadIdx.x;
    const int lane = tid & 31;
    const int wave = tid >> 5;
    const int wm   = (wave & 1) * 32;
    const int wn   = (wave >> 1) * 32;
    const int m0   = blockIdx.y * BM;
    const int n0   = blockIdx.x * BN;
    const int nk   = K / BK;

    f32x8 acc[2][2];
#pragma unroll
    for (int i = 0; i < 2; ++i)
#pragma unroll
        for (int j = 0; j < 2; ++j)
#pragma unroll
            for (int v = 0; v < 8; ++v) acc[i][j][v] = 0.f;

#if HAVE_TDM
    if (wave == 0) {
        tdm_issue_tile(A + (long)m0 * K, lds_off(&sA[0][0]), BM, K);
        tdm_issue_tile(W + (long)n0 * K, lds_off(&sB[0][0]), BN, K);
    }
#else
    const int ar = tid >> 2;
    const int ac = (tid & 3) * 8;
#endif

    for (int i = 0; i < nk; ++i) {
        const int cur = i & 1;
        const int kb  = i * BK;
#if HAVE_TDM
        if (wave == 0) {
            if (i + 1 < nk) {
                tdm_issue_tile(A + (long)m0 * K + kb + BK, lds_off(&sA[cur ^ 1][0]), BM, K);
                tdm_issue_tile(W + (long)n0 * K + kb + BK, lds_off(&sB[cur ^ 1][0]), BN, K);
                __builtin_amdgcn_s_wait_tensorcnt(2);  /* tile i's pair complete (in-order) */
            } else {
                __builtin_amdgcn_s_wait_tensorcnt(0);
            }
        }
        __syncthreads();
#else
        *(bf16x8*)(&sA[cur][ar * LDT + ac]) =
            *(const bf16x8*)(&A[(long)(m0 + ar) * K + kb + ac]);
#pragma unroll
        for (int c = 0; c < 2; ++c) {
            int idx = tid + c * 256;
            int br  = idx >> 2;
            int bc  = (idx & 3) * 8;
            *(bf16x8*)(&sB[cur][br * LDT + bc]) =
                *(const bf16x8*)(&W[(long)(n0 + br) * K + kb + bc]);
        }
        __syncthreads();
#endif

        bf16x16 afr[2], bfr[2];
        afr[0] = frag_a(sA[cur], wm);
        afr[1] = frag_a(sA[cur], wm + 16);
        bfr[0] = frag_b(sB[cur], wn);
        bfr[1] = frag_b(sB[cur], wn + 16);
#pragma unroll
        for (int ii = 0; ii < 2; ++ii)
#pragma unroll
            for (int jj = 0; jj < 2; ++jj)
                acc[ii][jj] = __builtin_amdgcn_wmma_f32_16x16x32_bf16(
                    false, afr[ii], false, bfr[jj], (short)0, acc[ii][jj], false, false);
        __syncthreads();
    }

    const int cn = lane & 15;
    const int cm = (lane & 16) ? 8 : 0;
#pragma unroll
    for (int i = 0; i < 2; ++i)
#pragma unroll
        for (int j = 0; j < 2; ++j) {
            int col = n0 + wn + j * 16 + cn;
#pragma unroll
            for (int v = 0; v < 8; ++v) {
                int   row = m0 + wm + i * 16 + cm + v;
                float val = acc[i][j][v];
                if (bias) val += bias[col];
                if (act == 1)  /* fast softplus: v_exp_f32 + v_log_f32 */
                    val = (val > 15.f) ? val : __logf(1.f + __expf(val));
                C[(long)row * N + col] = val;
            }
        }
}

/* ---------------- elementwise helpers ---------------- */
__global__ void cvt_f32_bf16(const float* __restrict__ in, bf16_t* __restrict__ out, long n) {
    long i = (long)blockIdx.x * blockDim.x + threadIdx.x;
    long s = (long)gridDim.x * blockDim.x;
    for (; i < n; i += s) out[i] = (bf16_t)in[i];
}

__global__ void cvt_pad_rows(const float* __restrict__ in, bf16_t* __restrict__ out,
                             int rows_in, int rows_out, int K) {
    long n = (long)rows_out * K;
    long i = (long)blockIdx.x * blockDim.x + threadIdx.x;
    long s = (long)gridDim.x * blockDim.x;
    for (; i < n; i += s) {
        int r = (int)(i / K);
        int k = (int)(i % K);
        out[i] = (r < rows_in) ? (bf16_t)in[(long)r * K + k] : (bf16_t)0.f;
    }
}

__global__ void extract_dtlo(const float* __restrict__ x_dbl, bf16_t* __restrict__ out) {
    long n = (long)MROWS * RRANK;
    long i = (long)blockIdx.x * blockDim.x + threadIdx.x;
    long s = (long)gridDim.x * blockDim.x;
    for (; i < n; i += s) {
        long row = i >> 6;
        int  c   = (int)(i & 63);
        out[i] = (bf16_t)x_dbl[row * XPAD + c];
    }
}

/* depthwise causal conv width-4 + bias + SiLU:  xz[:, :DINNER] -> u */
__global__ void conv_silu(const float* __restrict__ xz, const float* __restrict__ cw,
                          const float* __restrict__ cb, float* __restrict__ u) {
    long total = (long)MROWS * DINNER;
    long idx = (long)blockIdx.x * blockDim.x + threadIdx.x;
    if (idx >= total) return;
    int  d   = (int)(idx % DINNER);
    long row = idx / DINNER;
    int  l   = (int)(row % SEQ);
    long bb  = (row - l) * (long)(2 * DINNER);
    float acc = cb[d];
#pragma unroll
    for (int j = 0; j < CONVW; ++j) {
        int ll = l - (CONVW - 1) + j;
        if (ll >= 0) acc += xz[bb + (long)ll * (2 * DINNER) + d] * cw[d * CONVW + j];
    }
    u[idx] = acc / (1.f + __expf(-acc));
}

/* selective scan: 16 n-lanes per d channel, sequential in t */
__global__ void __launch_bounds__(256)
scan_kernel(const float* __restrict__ delta, const float* __restrict__ u,
            const float* __restrict__ x_dbl, const float* __restrict__ A_log,
            float* __restrict__ y) {
    int n    = threadIdx.x & 15;
    int dloc = threadIdx.x >> 4;
    int d    = blockIdx.x * 16 + dloc;
    int b    = blockIdx.y;
    float An = -__expf(A_log[d * NSTATE + n]);
    float h  = 0.f;
    const long baseDU = (long)b * SEQ * DINNER + d;
    const long baseX  = (long)b * SEQ * XPAD;
    for (int t = 0; t < SEQ; ++t) {
        float dt = delta[baseDU + (long)t * DINNER];
        float ut = u[baseDU + (long)t * DINNER];
        float Bn = x_dbl[baseX + t * XPAD + RRANK + n];
        float Cn = x_dbl[baseX + t * XPAD + RRANK + NSTATE + n];
        float dA = __expf(dt * An);
        h = dA * h + (dt * ut) * Bn;
        float yv = h * Cn;
        yv += __shfl_xor(yv, 1, 16);
        yv += __shfl_xor(yv, 2, 16);
        yv += __shfl_xor(yv, 4, 16);
        yv += __shfl_xor(yv, 8, 16);
        if (n == 0) y[baseDU + (long)t * DINNER] = yv;
    }
}

/* (y + u*D) * silu(z)  ->  bf16 for out_proj */
__global__ void gate_kernel(const float* __restrict__ y, const float* __restrict__ u,
                            const float* __restrict__ xz, const float* __restrict__ Dp,
                            bf16_t* __restrict__ yg) {
    long total = (long)MROWS * DINNER;
    long idx = (long)blockIdx.x * blockDim.x + threadIdx.x;
    if (idx >= total) return;
    int  d   = (int)(idx % DINNER);
    long row = idx / DINNER;
    float z  = xz[row * (2 * DINNER) + DINNER + d];
    float sz = z / (1.f + __expf(-z));
    yg[idx]  = (bf16_t)((y[idx] + u[idx] * Dp[d]) * sz);
}

/* ---------------- launch ---------------- */
extern "C" void kernel_launch(void* const* d_in, const int* in_sizes, int n_in,
                              void* d_out, int out_size, void* d_ws, size_t ws_size,
                              hipStream_t stream) {
    const float* hs    = (const float*)d_in[0];
    const float* w_in  = (const float*)d_in[1];
    const float* cw    = (const float*)d_in[2];
    const float* cb    = (const float*)d_in[3];
    const float* w_x   = (const float*)d_in[4];
    const float* w_dt  = (const float*)d_in[5];
    const float* b_dt  = (const float*)d_in[6];
    const float* A_log = (const float*)d_in[7];
    const float* Dp    = (const float*)d_in[8];
    const float* w_out = (const float*)d_in[9];
    float* out = (float*)d_out;

    char*  ws  = (char*)d_ws;
    size_t off = 0;
    auto alloc = [&](size_t bytes) -> void* {
        off = (off + 255) & ~(size_t)255;
        void* p = ws + off;
        off += bytes;
        return p;
    };

    bf16_t* hs_bf    = (bf16_t*)alloc((size_t)MROWS * DMODEL * 2);
    bf16_t* w_in_bf  = (bf16_t*)alloc((size_t)(2 * DINNER) * DMODEL * 2);
    float*  xz       = (float*) alloc((size_t)MROWS * (2 * DINNER) * 4);
    float*  u        = (float*) alloc((size_t)MROWS * DINNER * 4);
    bf16_t* u_bf     = (bf16_t*)alloc((size_t)MROWS * DINNER * 2);
    bf16_t* w_x_bf   = (bf16_t*)alloc((size_t)XPAD * DINNER * 2);
    float*  x_dbl    = (float*) alloc((size_t)MROWS * XPAD * 4);
    bf16_t* dtlo_bf  = (bf16_t*)alloc((size_t)MROWS * RRANK * 2);
    bf16_t* w_dt_bf  = (bf16_t*)alloc((size_t)DINNER * RRANK * 2);
    float*  delta    = (float*) alloc((size_t)MROWS * DINNER * 4);
    float*  yscan    = (float*) alloc((size_t)MROWS * DINNER * 4);
    bf16_t* yg_bf    = (bf16_t*)alloc((size_t)MROWS * DINNER * 2);
    bf16_t* w_out_bf = (bf16_t*)alloc((size_t)DMODEL * DINNER * 2);

    const int TB = 256;
    auto blocks = [&](long n) { return (unsigned)((n + TB - 1) / TB); };

    long n_hs = (long)MROWS * DMODEL;
    cvt_f32_bf16<<<blocks(n_hs), TB, 0, stream>>>(hs, hs_bf, n_hs);
    long n_wi = (long)(2 * DINNER) * DMODEL;
    cvt_f32_bf16<<<blocks(n_wi), TB, 0, stream>>>(w_in, w_in_bf, n_wi);

    gemm_bf16_nt<<<dim3((2 * DINNER) / BN, MROWS / BM), TB, 0, stream>>>(
        hs_bf, w_in_bf, xz, nullptr, MROWS, 2 * DINNER, DMODEL, 0);

    conv_silu<<<blocks((long)MROWS * DINNER), TB, 0, stream>>>(xz, cw, cb, u);

    cvt_f32_bf16<<<blocks((long)MROWS * DINNER), TB, 0, stream>>>(u, u_bf, (long)MROWS * DINNER);
    cvt_pad_rows<<<blocks((long)XPAD * DINNER), TB, 0, stream>>>(w_x, w_x_bf, RRANK + 2 * NSTATE, XPAD, DINNER);
    gemm_bf16_nt<<<dim3(XPAD / BN, MROWS / BM), TB, 0, stream>>>(
        u_bf, w_x_bf, x_dbl, nullptr, MROWS, XPAD, DINNER, 0);

    extract_dtlo<<<blocks((long)MROWS * RRANK), TB, 0, stream>>>(x_dbl, dtlo_bf);
    cvt_f32_bf16<<<blocks((long)DINNER * RRANK), TB, 0, stream>>>(w_dt, w_dt_bf, (long)DINNER * RRANK);
    gemm_bf16_nt<<<dim3(DINNER / BN, MROWS / BM), TB, 0, stream>>>(
        dtlo_bf, w_dt_bf, delta, b_dt, MROWS, DINNER, RRANK, 1);

    scan_kernel<<<dim3(DINNER / 16, BATCH), TB, 0, stream>>>(delta, u, x_dbl, A_log, yscan);

    gate_kernel<<<blocks((long)MROWS * DINNER), TB, 0, stream>>>(yscan, u, xz, Dp, yg_bf);

    cvt_f32_bf16<<<blocks((long)DMODEL * DINNER), TB, 0, stream>>>(w_out, w_out_bf, (long)DMODEL * DINNER);
    gemm_bf16_nt<<<dim3(DMODEL / BN, MROWS / BM), TB, 0, stream>>>(
        yg_bf, w_out_bf, out, nullptr, MROWS, DMODEL, DINNER, 0);
}
